// TokenRoutedMLPTriton_18047452578003
// MI455X (gfx1250) — matrix operand order repaired and checked by reading, not compiled, not run on hardware
//
#include <hip/hip_runtime.h>

#define T_TOKENS 4096
#define H_DIM    2048
#define EI_DIM   2048
#define E_NUM    4
#define VOCAB    100000
#define BM       32
#define BN       128
#define BK       64
#define LDSK     (BK + 8)                   // 72 elems = 144 B row stride (16B-aligned)
#define MAXROWS  (T_TOKENS + E_NUM * BM)    // 4224 padded rows worst case

typedef __attribute__((ext_vector_type(16))) __bf16 v16bf;
typedef __attribute__((ext_vector_type(8)))  __bf16 v8bf;
typedef __attribute__((ext_vector_type(8)))  float  v8f;
typedef __attribute__((ext_vector_type(4)))  int    v4i;

__device__ __forceinline__ __bf16 f2bf(float f) {
    unsigned u = __float_as_uint(f);
    unsigned r = (u + 0x7FFFu + ((u >> 16) & 1u)) >> 16;   // RNE
    unsigned short h = (unsigned short)r;
    return *reinterpret_cast<__bf16*>(&h);
}

// ---- async global->LDS copy (ASYNCcnt path), with safe fallback -------------
#if __has_builtin(__builtin_amdgcn_global_load_async_to_lds_b128)
#define HAVE_ASYNC 1
#define ASYNC_CP_B128(g, l)                                                    \
    __builtin_amdgcn_global_load_async_to_lds_b128(                            \
        (__attribute__((address_space(1))) v4i*)(unsigned long long)(g),       \
        (__attribute__((address_space(3))) v4i*)(unsigned)(unsigned long long)(l), \
        0, 0)
#if __has_builtin(__builtin_amdgcn_s_wait_asynccnt)
#define ASYNC_WAIT_N(n) __builtin_amdgcn_s_wait_asynccnt(n)
#else
#define ASYNC_WAIT_N(n) asm volatile("s_wait_asynccnt %0" :: "i"(n) : "memory")
#endif
#else
#define HAVE_ASYNC 0
#define ASYNC_CP_B128(g, l) do { *(uint4*)(l) = *(const uint4*)(g); } while (0)
#define ASYNC_WAIT_N(n) do {} while (0)
#endif

// ---- fragment builders ------------------------------------------------------
// A operand (16x32 bf16): lane = M row; elems 0..7 = K kb..kb+7, 8..15 = kb+16..kb+23
__device__ __forceinline__ v16bf load_afrag(const __bf16* rowbase, int ksub, int lhalf) {
    int kb = ksub * 32 + lhalf * 8;
    v8bf lo = *(const v8bf*)(rowbase + kb);
    v8bf hi = *(const v8bf*)(rowbase + kb + 16);
    v16bf r;
#pragma unroll
    for (int i = 0; i < 8; ++i) { r[i] = lo[i]; r[8 + i] = hi[i]; }
    return r;
}
// B operand (32x16 bf16): lane = N col; elems 0..7 = K kb..kb+7, 8..15 = kb+16..kb+23
__device__ __forceinline__ v16bf load_bfrag(const __bf16* rowbase, int ksub, int lhalf) {
    int ks = ksub * 32 + lhalf * 8;
    v8bf lo = *(const v8bf*)(rowbase + ks);
    v8bf hi = *(const v8bf*)(rowbase + ks + 16);
    v16bf r;
#pragma unroll
    for (int i = 0; i < 8; ++i) { r[i] = lo[i]; r[8 + i] = hi[i]; }
    return r;
}

// ---- fused transpose + fp32->bf16 weight conversion ------------------------
// src: [E][KD][ND] fp32  ->  dst: [E][ND][KD] bf16  (runs once per launch)
__global__ void __launch_bounds__(256)
cvtT_kernel(const float* __restrict__ src, __bf16* __restrict__ dst, int KD, int ND) {
    __shared__ float tile[32][33];
    int e  = blockIdx.z;
    int k0 = blockIdx.y * 32;
    int n0 = blockIdx.x * 32;
    int tr = threadIdx.x >> 3;              // 0..31
    int tc = (threadIdx.x & 7) * 4;         // 0..28
    const float* s = src + ((size_t)e * KD + k0) * ND + n0;
    float4 v = *(const float4*)(s + (size_t)tr * ND + tc);
    tile[tr][tc + 0] = v.x; tile[tr][tc + 1] = v.y;
    tile[tr][tc + 2] = v.z; tile[tr][tc + 3] = v.w;
    __syncthreads();
    __bf16* d = dst + ((size_t)e * ND + n0 + tr) * KD + k0;
    __bf16 o[4] = { f2bf(tile[tc + 0][tr]), f2bf(tile[tc + 1][tr]),
                    f2bf(tile[tc + 2][tr]), f2bf(tile[tc + 3][tr]) };
    *(uint2*)(d + tc) = *(const uint2*)o;
}

// ---- routing ----------------------------------------------------------------
__global__ void __launch_bounds__(256)
count_kernel(const int* __restrict__ tok, int* __restrict__ counts) {
    int t = blockIdx.x * 256 + threadIdx.x;
    if (t >= T_TOKENS) return;
    int id = tok[t];
    id = id < 0 ? 0 : (id >= VOCAB ? VOCAB - 1 : id);
    atomicAdd(&counts[id % E_NUM], 1);
}

__global__ void offsets_kernel(const int* __restrict__ counts, int* __restrict__ offs) {
    if (threadIdx.x == 0 && blockIdx.x == 0) {
        int acc = 0;
        for (int e = 0; e < E_NUM; ++e) {
            offs[e] = acc;
            acc += ((counts[e] + BM - 1) / BM) * BM;
        }
        offs[E_NUM] = acc;
    }
}

__global__ void __launch_bounds__(256)
pad_kernel(const int* __restrict__ counts, const int* __restrict__ offs,
           int* __restrict__ row_map) {
    int r = blockIdx.x * 256 + threadIdx.x;
    if (r >= MAXROWS) return;
    int total = offs[E_NUM];
    if (r >= total) { row_map[r] = -1; return; }
    int e = 0;
    for (int i = 1; i < E_NUM; ++i) if (r >= offs[i]) e = i;
    if (r >= offs[e] + counts[e]) row_map[r] = -1;
}

__global__ void __launch_bounds__(256)
gather_kernel(const float* __restrict__ x, const int* __restrict__ tok,
              const int* __restrict__ offs, int* __restrict__ cursors,
              int* __restrict__ row_map, __bf16* __restrict__ Xg) {
    __shared__ int sr;
    int t = blockIdx.x;
    if (threadIdx.x == 0) {
        int id = tok[t];
        id = id < 0 ? 0 : (id >= VOCAB ? VOCAB - 1 : id);
        int e = id % E_NUM;
        int r = offs[e] + atomicAdd(&cursors[e], 1);
        row_map[r] = t;
        sr = r;
    }
    __syncthreads();
    int r = sr;
    const float4* src = (const float4*)(x + (size_t)t * H_DIM);
    __bf16* dst = Xg + (size_t)r * H_DIM;
    for (int i = threadIdx.x; i < H_DIM / 4; i += 256) {
        float4 v = src[i];
        __bf16 o[4] = { f2bf(v.x), f2bf(v.y), f2bf(v.z), f2bf(v.w) };
        *(uint2*)(dst + i * 4) = *(const uint2*)o;
    }
}

// ---- GEMM1: gate+up fused, SwiGLU epilogue ----------------------------------
// Tile 32x128, K-step 64; double-buffered async pipeline; 8 waves 2x4;
// each wave: 16 rows x 32 cols x {gate, up} = 8 WMMA per K-step.
__global__ void __launch_bounds__(256)
gemm_gateup_kernel(const __bf16* __restrict__ Xg,
                   const __bf16* __restrict__ WgT,  // [E][EI][H] bf16 (K contig)
                   const __bf16* __restrict__ WuT,
                   __bf16* __restrict__ inter,
                   const int* __restrict__ offs) {
    __shared__ __align__(16) __bf16 As[2][BM][LDSK];
    __shared__ __align__(16) __bf16 Bg[2][BN][LDSK];
    __shared__ __align__(16) __bf16 Bu[2][BN][LDSK];

    int rb = blockIdx.y * BM;
    int total = offs[E_NUM];
    if (rb >= total) return;                 // uniform per block
    int e = 0;
    for (int i = 1; i < E_NUM; ++i) if (rb >= offs[i]) e = i;
    int n0 = blockIdx.x * BN;

    int tid   = threadIdx.x;
    int lane  = tid & 31;
    int wave  = tid >> 5;
    int mr    = wave >> 2;                   // 0..1 (row half)
    int nc    = wave & 3;                    // 0..3 (32-col group)
    int lhalf = lane >> 4;
    int l15   = lane & 15;

    const __bf16* arow = Xg  + (size_t)rb * H_DIM;
    const __bf16* gw   = WgT + ((size_t)e * EI_DIM + n0) * H_DIM;
    const __bf16* uw   = WuT + ((size_t)e * EI_DIM + n0) * H_DIM;

    v8f accg[2] = {{}, {}};
    v8f accu[2] = {{}, {}};

    int arowl = tid >> 3, akk = (tid & 7) * 8;      // A copy slot (1 x b128)

    // 9 async copy instructions per thread per tile (1 A + 4x2 B)
    auto copy_tiles = [&](int k0c, int b) {
        ASYNC_CP_B128(arow + (size_t)arowl * H_DIM + k0c + akk, &As[b][arowl][akk]);
#pragma unroll
        for (int j = 0; j < 4; ++j) {
            int c = tid + j * 256;
            int row = c >> 3, kk = (c & 7) * 8;
            ASYNC_CP_B128(gw + (size_t)row * H_DIM + k0c + kk, &Bg[b][row][kk]);
            ASYNC_CP_B128(uw + (size_t)row * H_DIM + k0c + kk, &Bu[b][row][kk]);
        }
    };

    copy_tiles(0, 0);
    int cur = 0;
    for (int k0 = 0; k0 < H_DIM; k0 += BK) {
        int nxt = cur ^ 1;
        if (k0 + BK < H_DIM) {               // prefetch next tile, keep 9 in flight
            copy_tiles(k0 + BK, nxt);
            ASYNC_WAIT_N(9);                 // in-order: current tile's 9 are done
        } else {
            ASYNC_WAIT_N(0);
        }
        __syncthreads();

        const __bf16* ab = &As[cur][mr * 16 + l15][0];
        v16bf a0 = load_afrag(ab, 0, lhalf);
        v16bf a1 = load_afrag(ab, 1, lhalf);
#pragma unroll
        for (int s = 0; s < 2; ++s) {
            const __bf16* gb = &Bg[cur][nc * 32 + s * 16 + l15][0];
            const __bf16* ub = &Bu[cur][nc * 32 + s * 16 + l15][0];
            v16bf bg0 = load_bfrag(gb, 0, lhalf);
            v16bf bg1 = load_bfrag(gb, 1, lhalf);
            v16bf bu0 = load_bfrag(ub, 0, lhalf);
            v16bf bu1 = load_bfrag(ub, 1, lhalf);
            accg[s] = __builtin_amdgcn_wmma_f32_16x16x32_bf16(false, a0, false, bg0,
                                                              (short)0, accg[s], false, false);
            accg[s] = __builtin_amdgcn_wmma_f32_16x16x32_bf16(false, a1, false, bg1,
                                                              (short)0, accg[s], false, false);
            accu[s] = __builtin_amdgcn_wmma_f32_16x16x32_bf16(false, a0, false, bu0,
                                                              (short)0, accu[s], false, false);
            accu[s] = __builtin_amdgcn_wmma_f32_16x16x32_bf16(false, a1, false, bu1,
                                                              (short)0, accu[s], false, false);
        }
        __syncthreads();                      // all reads of 'cur' done before overwrite
        cur = nxt;
    }

    // Epilogue: inter = silu(gate) * up, bf16
#pragma unroll
    for (int s = 0; s < 2; ++s) {
        int col = n0 + nc * 32 + s * 16 + l15;
#pragma unroll
        for (int i = 0; i < 8; ++i) {
            int row = rb + mr * 16 + lhalf * 8 + i;
            float g = accg[s][i];
            float sig = __fdividef(g, 1.0f + __expf(-g));
            inter[(size_t)row * EI_DIM + col] = f2bf(sig * accu[s][i]);
        }
    }
}

// ---- GEMM2: down projection + scatter to token order ------------------------
__global__ void __launch_bounds__(256)
gemm_down_kernel(const __bf16* __restrict__ inter,
                 const __bf16* __restrict__ WdT,   // [E][H][EI] bf16 (K contig)
                 float* __restrict__ out,
                 const int* __restrict__ offs,
                 const int* __restrict__ row_map) {
    __shared__ __align__(16) __bf16 As[2][BM][LDSK];
    __shared__ __align__(16) __bf16 Bd[2][BN][LDSK];

    int rb = blockIdx.y * BM;
    int total = offs[E_NUM];
    if (rb >= total) return;
    int e = 0;
    for (int i = 1; i < E_NUM; ++i) if (rb >= offs[i]) e = i;
    int n0 = blockIdx.x * BN;

    int tid   = threadIdx.x;
    int lane  = tid & 31;
    int wave  = tid >> 5;
    int mr    = wave >> 2;
    int nc    = wave & 3;
    int lhalf = lane >> 4;
    int l15   = lane & 15;

    const __bf16* arow = inter + (size_t)rb * EI_DIM;
    const __bf16* dw   = WdT + ((size_t)e * H_DIM + n0) * EI_DIM;

    v8f acc[2] = {{}, {}};
    int arowl = tid >> 3, akk = (tid & 7) * 8;

    // 5 async copy instructions per thread per tile (1 A + 4 B)
    auto copy_tiles = [&](int k0c, int b) {
        ASYNC_CP_B128(arow + (size_t)arowl * EI_DIM + k0c + akk, &As[b][arowl][akk]);
#pragma unroll
        for (int j = 0; j < 4; ++j) {
            int c = tid + j * 256;
            int row = c >> 3, kk = (c & 7) * 8;
            ASYNC_CP_B128(dw + (size_t)row * EI_DIM + k0c + kk, &Bd[b][row][kk]);
        }
    };

    copy_tiles(0, 0);
    int cur = 0;
    for (int k0 = 0; k0 < EI_DIM; k0 += BK) {
        int nxt = cur ^ 1;
        if (k0 + BK < EI_DIM) {
            copy_tiles(k0 + BK, nxt);
            ASYNC_WAIT_N(5);
        } else {
            ASYNC_WAIT_N(0);
        }
        __syncthreads();

        const __bf16* ab = &As[cur][mr * 16 + l15][0];
        v16bf a0 = load_afrag(ab, 0, lhalf);
        v16bf a1 = load_afrag(ab, 1, lhalf);
#pragma unroll
        for (int s = 0; s < 2; ++s) {
            const __bf16* bb = &Bd[cur][nc * 32 + s * 16 + l15][0];
            v16bf b0 = load_bfrag(bb, 0, lhalf);
            v16bf b1 = load_bfrag(bb, 1, lhalf);
            acc[s] = __builtin_amdgcn_wmma_f32_16x16x32_bf16(false, a0, false, b0,
                                                             (short)0, acc[s], false, false);
            acc[s] = __builtin_amdgcn_wmma_f32_16x16x32_bf16(false, a1, false, b1,
                                                             (short)0, acc[s], false, false);
        }
        __syncthreads();
        cur = nxt;
    }

    // Scatter fp32 rows back to token order; pad rows (map == -1) skipped.
#pragma unroll
    for (int s = 0; s < 2; ++s) {
        int col = n0 + nc * 32 + s * 16 + l15;
#pragma unroll
        for (int i = 0; i < 8; ++i) {
            int row = rb + mr * 16 + lhalf * 8 + i;
            int t = row_map[row];
            if (t >= 0) out[(size_t)t * H_DIM + col] = acc[s][i];
        }
    }
}

// -----------------------------------------------------------------------------
extern "C" void kernel_launch(void* const* d_in, const int* in_sizes, int n_in,
                              void* d_out, int out_size, void* d_ws, size_t ws_size,
                              hipStream_t stream) {
    const float* x   = (const float*)d_in[0];   // [T, H]
    const int*   tok = (const int*)d_in[1];     // [T]
    const float* gw  = (const float*)d_in[2];   // [E, H, EI]
    const float* uw  = (const float*)d_in[3];   // [E, H, EI]
    const float* dw  = (const float*)d_in[4];   // [E, EI, H]
    float* out = (float*)d_out;

    char* ws = (char*)d_ws;
    int* counts  = (int*)ws;
    int* cursors = counts + 8;
    int* offs    = counts + 16;
    int* row_map = counts + 32;

    const size_t WELEMS = (size_t)E_NUM * H_DIM * EI_DIM;
    __bf16* Xg    = (__bf16*)(ws + 65536);
    __bf16* inter = (__bf16*)(ws + 65536 + (size_t)MAXROWS * H_DIM * 2);
    __bf16* WgT   = (__bf16*)(ws + 65536 + (size_t)MAXROWS * H_DIM * 2
                                         + (size_t)MAXROWS * EI_DIM * 2);
    __bf16* WuT   = WgT + WELEMS;
    __bf16* WdT   = WuT + WELEMS;

    (void)hipMemsetAsync(ws, 0, 64, stream);    // counts + cursors

    // one-time fp32->bf16 transpose-convert; bf16 weights (~100 MB) stay in L2
    dim3 gc(EI_DIM / 32, H_DIM / 32, E_NUM);
    cvtT_kernel<<<gc, 256, 0, stream>>>(gw, WgT, H_DIM, EI_DIM);
    cvtT_kernel<<<gc, 256, 0, stream>>>(uw, WuT, H_DIM, EI_DIM);
    dim3 gd(H_DIM / 32, EI_DIM / 32, E_NUM);
    cvtT_kernel<<<gd, 256, 0, stream>>>(dw, WdT, EI_DIM, H_DIM);

    // routing
    count_kernel<<<T_TOKENS / 256, 256, 0, stream>>>(tok, counts);
    offsets_kernel<<<1, 32, 0, stream>>>(counts, offs);
    pad_kernel<<<(MAXROWS + 255) / 256, 256, 0, stream>>>(counts, offs, row_map);
    gather_kernel<<<T_TOKENS, 256, 0, stream>>>(x, tok, offs, cursors, row_map, Xg);

    // grouped GEMMs
    dim3 g1(EI_DIM / BN, MAXROWS / BM);
    gemm_gateup_kernel<<<g1, 256, 0, stream>>>(Xg, WgT, WuT, inter, offs);
    dim3 g2(H_DIM / BN, MAXROWS / BM);
    gemm_down_kernel<<<g2, 256, 0, stream>>>(inter, WdT, out, offs, row_map);
}
